// BatchCRFLoss_13915694039462
// MI455X (gfx1250) — matrix operand order repaired and checked by reading, not compiled, not run on hardware
//
#include <hip/hip_runtime.h>

#define Nn 64
#define Cc 64
#define Ll 8192

#define LOG2E 1.44269504088896340736f
#define LN2   0.69314718055994530942f

typedef __attribute__((ext_vector_type(16))) _Float16 v16h;
typedef __attribute__((ext_vector_type(8)))  float    v8f;

// Workgroup barrier that only drains LDS (DScnt), leaving global prefetch
// loads in flight (unlike __syncthreads, which also drains LOADcnt).
#define WG_BARRIER_LDS()                                                    \
  asm volatile("s_wait_dscnt 0x0\n\t"                                       \
               "s_barrier_signal -1\n\t"                                    \
               "s_barrier_wait -1" ::: "memory")

// ---------------------------------------------------------------------------
// Kernel 1: gold score per batch row.
// gold[n] = sum_t obs[n, tgt[n,t], t] + sum_{t<L-1} T[tgt[n,t], tgt[n,t+1]]
// ---------------------------------------------------------------------------
__global__ __launch_bounds__(256) void crf_gold_kernel(
    const float* __restrict__ obs, const float* __restrict__ trans,
    const int* __restrict__ tgt, float* __restrict__ gold_out) {
  const int n = blockIdx.x;
  const int tid = threadIdx.x;
  float s = 0.0f;
  for (int t = tid; t < Ll; t += 256) {
    int c = tgt[n * Ll + t];
    s += obs[(n * Cc + c) * Ll + t];
    if (t + 1 < Ll) {
      int c2 = tgt[n * Ll + t + 1];
      s += trans[c * Cc + c2];
    }
  }
  __shared__ float red[256];
  red[tid] = s;
  __syncthreads();
  for (int off = 128; off >= 1; off >>= 1) {
    if (tid < off) red[tid] += red[tid + off];
    __syncthreads();
  }
  if (tid == 0) gold_out[n] = red[0];
}

// ---------------------------------------------------------------------------
// Kernel 2: forward scan with linear-domain propagation.
// 4 blocks (16 batch rows each) x 4 waves (one 16x16 output tile each).
// LDS holds a_t = 2^(beta_t - P_t) (f32); P_t is an exact per-row integer
// scale. Per step: a' = (A x B) * 2^(e*log2e) * 2^(-k-5), P += k+5, where
// k = max of the frexp exponents of 2 row samples (broadcast loads,
// row-consistent). Raw emissions are rotated one iteration ahead, and the
// loop barrier drains only DScnt, so prefetch loads stay in flight across
// it; logs run once at the very end.
// ---------------------------------------------------------------------------
__global__ __launch_bounds__(128, 1) void crf_scan_kernel(
    const float* __restrict__ obs, const float* __restrict__ trans,
    float* __restrict__ logZ_out) {
  __shared__ float lds_a[2][16][72];   // double-buffered linear-domain tile
  __shared__ float lds_pmax[2][16][4]; // finale only
  __shared__ float lds_max[16];        // finale only

  const int tid = threadIdx.x;
  const int w   = tid >> 5;      // wave id == output column tile (0..3)
  const int l   = tid & 31;      // lane in wave32
  const int ln  = l & 15;        // lane within half
  const int hl  = l >> 4;        // lane half
  const int M0  = hl * 8;        // D-layout row base for this lane
  const int n0  = blockIdx.x * 16;

  // ---- Preload B fragments: B_ij = e^{T_ij}, columns [16w, 16w+16).
  // B layout: VGPR v, lane l: N = l%16, K = 32*kc + 16*hl + i (h[i], i=0..15).
  v16h bfrag0, bfrag1;
  {
    const int col = 16 * w + ln;
#pragma unroll
    for (int i = 0; i < 16; ++i) {
      bfrag0[i] = (_Float16)__expf(trans[(16 * hl + i) * Cc + col]);
      bfrag1[i] = (_Float16)__expf(trans[(32 + 16 * hl + i) * Cc + col]);
    }
  }

  // ---- Per-lane element offsets into obs for this wave's D tile.
  // D element r: row M = M0+r (batch n0+M), col Ng = 16w+ln (state).
  // voff[r] - voff[0] = r * Cc * Ll  (r*2MB in bytes: rows 0..3 fit the
  // signed 24-bit instruction offset from one base, rows 4..7 from another).
  int voff[8];
#pragma unroll
  for (int r = 0; r < 8; ++r)
    voff[r] = ((n0 + M0 + r) * Cc + (16 * w + ln)) * Ll;

  // ---- Init: a_0 = 2^(e_0*log2e) (P_0 = 0); prime raw e with t=1.
  float e[8];
  int   P[8];
#pragma unroll
  for (int r = 0; r < 8; ++r) {
    lds_a[0][M0 + r][16 * w + ln] =
        __builtin_amdgcn_exp2f(obs[voff[r]] * LOG2E);
    e[r] = obs[voff[r] + 1];
    P[r] = 0;
  }
  v8f aw;  // this lane's last-produced a values (for the finale)
  __syncthreads();

  // ---- Sequential forward recursion.
  for (int t = 1; t < Ll; ++t) {
    const int rp = (t - 1) & 1;
    const int wp = t & 1;

    // Exponentiate this step's emissions (loaded last iteration -- already
    // in registers, so no load wait here).
    float x[8];
#pragma unroll
    for (int r = 0; r < 8; ++r)
      x[r] = __builtin_amdgcn_exp2f(e[r] * LOG2E);

    // Per-row power-of-two renormalizer: integer max of the frexp
    // exponents of 2 broadcast samples (identical in every lane ->
    // row-consistent, exact scaling, exact integer bookkeeping).
    float xu[8];
#pragma unroll
    for (int r = 0; r < 8; ++r) {
      int k0 = __builtin_amdgcn_frexp_expf(lds_a[rp][M0 + r][0]);
      int k1 = __builtin_amdgcn_frexp_expf(lds_a[rp][M0 + r][32]);
      int k = (k0 > k1) ? k0 : k1;
      xu[r] = __builtin_amdgcn_ldexpf(x[r], -k - 5);
      P[r] += k + 5;
    }

    // A fragments: straight load + convert (values already linear-domain).
    // A layout: lane l -> row ln; h[i]: K = 32*kc + 16*hl + i (i=0..15).
    v16h afrag0, afrag1;
#pragma unroll
    for (int i = 0; i < 8; ++i) {
      const int cb = 16 * hl;
      afrag0[i]     = (_Float16)lds_a[rp][ln][cb + i];
      afrag0[i + 8] = (_Float16)lds_a[rp][ln][cb + 8 + i];
      afrag1[i]     = (_Float16)lds_a[rp][ln][32 + cb + i];
      afrag1[i + 8] = (_Float16)lds_a[rp][ln][32 + cb + 8 + i];
    }

    // Prefetch next step's raw emissions (consumed next iteration; clamped
    // on the last step). Two base pointers + immediate offsets r*2MB.
    float en[8];
    {
      const int tn = (t + 1 < Ll) ? t + 1 : t;
      const float* q0 = obs + (voff[0] + tn);
      const float* q4 = obs + (voff[4] + tn);
#pragma unroll
      for (int i = 0; i < 4; ++i) {
        en[i]     = q0[i * (Cc * Ll)];
        en[i + 4] = q4[i * (Cc * Ll)];
      }
    }

    // GEMM: w_j = sum_i a_i * e^{T_ij}, K=64 as 2x K=32.
    v8f cz = {};
    v8f ws = __builtin_amdgcn_wmma_f32_16x16x32_f16(
        false, afrag0, false, bfrag0, (short)0, cz, false, false);
    ws = __builtin_amdgcn_wmma_f32_16x16x32_f16(
        false, afrag1, false, bfrag1, (short)0, ws, false, false);

    // Produce a_t = ws * 2^(e*log2e - k - 5); clamp keeps the later
    // f16 conversion finite even in astronomically unlikely tails.
#pragma unroll
    for (int r = 0; r < 8; ++r) {
      float a = fminf(ws[r] * xu[r], 60000.0f);
      aw[r] = a;
      lds_a[wp][M0 + r][16 * w + ln] = a;
    }

    // LDS-only barrier: stores visible WGP-wide, global prefetch unharmed.
    WG_BARRIER_LDS();

#pragma unroll
    for (int r = 0; r < 8; ++r) e[r] = en[r];
  }

  // ---- Reconstruct beta = log2(a) + P once, then exact logsumexp finale.
  v8f acc;
#pragma unroll
  for (int r = 0; r < 8; ++r)
    acc[r] = __builtin_amdgcn_logf(aw[r]) + (float)P[r];

  {
    v8f m = acc;
#pragma unroll
    for (int s = 1; s < 16; s <<= 1) {
#pragma unroll
      for (int r = 0; r < 8; ++r)
        m[r] = fmaxf(m[r], __shfl_xor(m[r], s, 32));
    }
    if (ln == 0) {
#pragma unroll
      for (int r = 0; r < 8; ++r) lds_pmax[0][M0 + r][w] = m[r];
    }
    __syncthreads();
    if (tid < 16) {
      float mm = fmaxf(fmaxf(lds_pmax[0][tid][0], lds_pmax[0][tid][1]),
                       fmaxf(lds_pmax[0][tid][2], lds_pmax[0][tid][3]));
      lds_max[tid] = mm;
    }
    __syncthreads();
    v8f sum;
#pragma unroll
    for (int r = 0; r < 8; ++r)
      sum[r] = __builtin_amdgcn_exp2f(acc[r] - lds_max[M0 + r]);
#pragma unroll
    for (int s = 1; s < 16; s <<= 1) {
#pragma unroll
      for (int r = 0; r < 8; ++r) sum[r] += __shfl_xor(sum[r], s, 32);
    }
    if (ln == 0) {
#pragma unroll
      for (int r = 0; r < 8; ++r) lds_pmax[1][M0 + r][w] = sum[r];
    }
    __syncthreads();
    if (tid < 16) {
      float tot = lds_pmax[1][tid][0] + lds_pmax[1][tid][1] +
                  lds_pmax[1][tid][2] + lds_pmax[1][tid][3];
      logZ_out[n0 + tid] =
          LN2 * (__builtin_amdgcn_logf(tot) + lds_max[tid]);
    }
  }
}

// ---------------------------------------------------------------------------
// Kernel 3: deterministic final reduction. out = mean_n(logZ[n] - gold[n])
// ---------------------------------------------------------------------------
__global__ __launch_bounds__(32) void crf_final_kernel(
    const float* __restrict__ gold, const float* __restrict__ logZ,
    float* __restrict__ out) {
  if (threadIdx.x == 0) {
    float s = 0.0f;
    for (int n = 0; n < Nn; ++n) s += logZ[n] - gold[n];
    out[0] = s / (float)Nn;
  }
}

extern "C" void kernel_launch(void* const* d_in, const int* in_sizes, int n_in,
                              void* d_out, int out_size, void* d_ws, size_t ws_size,
                              hipStream_t stream) {
  const float* obs   = (const float*)d_in[0];  // (N, C, L) f32
  const float* trans = (const float*)d_in[1];  // (C, C)    f32
  const int*   tgt   = (const int*)d_in[2];    // (N, L)    int
  float* out  = (float*)d_out;                 // scalar f32
  float* gold = (float*)d_ws;                  // [64]
  float* logZ = gold + Nn;                     // [64]

  crf_gold_kernel<<<Nn, 256, 0, stream>>>(obs, trans, tgt, gold);
  crf_scan_kernel<<<Nn / 16, 128, 0, stream>>>(obs, trans, logZ);
  crf_final_kernel<<<1, 32, 0, stream>>>(gold, logZ, out);
}
